// AlignedQuestionEmbeddingLayer_69544110457489
// MI455X (gfx1250) — matrix operand order repaired and checked by reading, not compiled, not run on hardware
//
#include <hip/hip_runtime.h>
#include <hip/hip_bf16.h>

// ---------------- types ----------------
typedef __bf16 v16bf __attribute__((ext_vector_type(16)));
typedef float  v8f   __attribute__((ext_vector_type(8)));
typedef unsigned int u32x4 __attribute__((ext_vector_type(4)));

union FragB { v16bf v; u32x4 q[2]; };

static constexpr int Bn  = 8;
static constexpr int CTX = 4096;
static constexpr int QST = 512;
static constexpr int E   = 300;
static constexpr int EP  = 320;   // E padded to multiple of 32
static constexpr int H   = 128;

static constexpr int CSTR = 136;  // LDS row stride (elems) for ctx_logits tile
static constexpr int PSTR = 72;   // LDS row stride (elems) for P tile

// native bf16 convert (lowers to v_cvt_*bf16_f32 on gfx1250)
__device__ __forceinline__ unsigned short f2bf(float x) {
    __bf16 h = (__bf16)x;
    return __builtin_bit_cast(unsigned short, h);
}

__device__ __forceinline__ unsigned pack2bf(float a, float b) {
    return (unsigned)f2bf(a) | ((unsigned)f2bf(b) << 16);
}

__device__ __forceinline__ u32x4 pack8bf(const v8f& a) {
    u32x4 r;
    r.x = pack2bf(a[0], a[1]);
    r.y = pack2bf(a[2], a[3]);
    r.z = pack2bf(a[4], a[5]);
    r.w = pack2bf(a[6], a[7]);
    return r;
}

__device__ __forceinline__ v8f wmma_bf16(v16bf a, v16bf b, v8f c) {
    // D = A(16x32 bf16) x B(32x16 bf16) + C(16x16 f32)
    return __builtin_amdgcn_wmma_f32_16x16x32_bf16(
        false, a, false, b, (short)0, c, false, false);
}

// ---- phase 1: acc[8] = relu( A(16 rows x EP) @ W^T + bias ), all 128 cols ----
// Apad: bf16 row-major stride EP, pointing at this wave's 16-row block.
// Wpad: bf16 [H][EP].
static __device__ __forceinline__ void dense_relu_tile(
    const unsigned short* __restrict__ Apad,
    const unsigned short* __restrict__ Wpad,
    const float* __restrict__ bias,
    int lane, v8f acc[8])
{
    const int wl   = lane & 15;
    const int half = lane >> 4;
    v8f zf = {};
    #pragma unroll
    for (int t = 0; t < 8; ++t) acc[t] = zf;

    const unsigned short* arow = Apad + (size_t)wl * EP;
    #pragma unroll
    for (int k0 = 0; k0 < EP; k0 += 32) {
        FragB a;                       // A-frag: lane half picks K {0..15} vs {16..31}
        const int ab = k0 + half * 16;
        a.q[0] = *(const u32x4*)(arow + ab);
        a.q[1] = *(const u32x4*)(arow + ab + 8);
        #pragma unroll
        for (int t = 0; t < 8; ++t) {  // 8 N-tiles cover H=128
            FragB w;                   // B-frag: lane = col h, 16 contiguous K
            const unsigned short* wp =
                Wpad + ((size_t)(t * 16 + wl) * EP + k0 + half * 16);
            w.q[0] = *(const u32x4*)wp;
            w.q[1] = *(const u32x4*)(wp + 8);
            acc[t] = wmma_bf16(a.v, w.v, acc[t]);
        }
    }
    #pragma unroll
    for (int t = 0; t < 8; ++t) {
        const float bb = bias[t * 16 + wl];
        #pragma unroll
        for (int j = 0; j < 8; ++j) acc[t][j] = fmaxf(acc[t][j] + bb, 0.f);
    }
}

// ---------------- kernel A: fp32 -> bf16, pad E(300) -> EP(320) ----------------
__global__ __launch_bounds__(256) void convert_pad_kernel(
    const float* __restrict__ src, unsigned short* __restrict__ dst, int rows)
{
    int idx = blockIdx.x * 256 + threadIdx.x;
    if (idx >= rows * EP) return;
    int r = idx / EP, c = idx - r * EP;
    dst[idx] = (c < E) ? f2bf(src[(size_t)r * E + c]) : (unsigned short)0;
}

// ---------------- kernel B: qst_logits in two layouts ----------------
__global__ __launch_bounds__(256) void qst_logits_kernel(
    const unsigned short* __restrict__ qst_pad,   // [Bn][QST][EP] bf16
    const unsigned short* __restrict__ Wpad,      // [H][EP] bf16
    const float* __restrict__ bias,
    unsigned short* __restrict__ ql,              // [Bn][QST][H] bf16
    unsigned short* __restrict__ qlT)             // [Bn][H][QST] bf16
{
    const int wave = threadIdx.x >> 5, lane = threadIdx.x & 31;
    const int wl = lane & 15, half = lane >> 4;
    const int b   = blockIdx.x >> 2;              // 4 row-blocks of 128 per batch
    const int q0  = (blockIdx.x & 3) * 128 + wave * 16;

    v8f acc[8];
    dense_relu_tile(qst_pad + ((size_t)b * QST + q0) * EP, Wpad, bias, lane, acc);

    // ql[b][q][h]  (D layout scatter, b16 stores)
    #pragma unroll
    for (int t = 0; t < 8; ++t)
        #pragma unroll
        for (int j = 0; j < 8; ++j)
            ql[((size_t)b * QST + q0 + half * 8 + j) * H + t * 16 + wl] = f2bf(acc[t][j]);

    // qlT[b][h][q] (8 consecutive q per lane -> one b128 store)
    #pragma unroll
    for (int t = 0; t < 8; ++t) {
        u32x4 pk = pack8bf(acc[t]);
        *(u32x4*)(qlT + ((size_t)b * H + t * 16 + wl) * QST + q0 + half * 8) = pk;
    }
}

// ---------------- kernel C: fused dense+scores+softmax+PV ----------------
__global__ __launch_bounds__(256) void fused_attn_kernel(
    const unsigned short* __restrict__ ctx_pad,   // [Bn][CTX][EP] bf16
    const unsigned short* __restrict__ Wpad,      // [H][EP] bf16
    const float* __restrict__ bias,               // [H]
    const unsigned short* __restrict__ ql,        // [Bn][QST][H] bf16
    const unsigned short* __restrict__ qlT,       // [Bn][H][QST] bf16
    const int* __restrict__ qmask,                // [Bn][QST]
    float* __restrict__ out)                      // [Bn][CTX][H]
{
    __shared__ unsigned short ctxbuf[8 * 16 * CSTR];  // per-wave 16x128 ctx_logits (bf16)
    __shared__ unsigned short pbuf[8 * 16 * PSTR];    // per-wave 16x64 P chunk (bf16)

    const int wave = threadIdx.x >> 5, lane = threadIdx.x & 31;
    const int wl = lane & 15, half = lane >> 4;
    const int b  = blockIdx.x >> 5;                    // 32 row-blocks per batch
    const int m0 = (blockIdx.x & 31) * 128 + wave * 16;

    const unsigned short* qlb  = ql  + (size_t)b * QST * H;
    const unsigned short* qlTb = qlT + (size_t)b * H * QST;
    const int* mb = qmask + b * QST;
    __builtin_prefetch(qlb, 0, 1);
    __builtin_prefetch(qlTb, 0, 1);

    // ---- phase 1: ctx_logits (16 x 128) ----
    v8f acc[8];
    dense_relu_tile(ctx_pad + ((size_t)b * CTX + m0) * EP, Wpad, bias, lane, acc);

    unsigned short* cb = ctxbuf + wave * 16 * CSTR;
    #pragma unroll
    for (int t = 0; t < 8; ++t)
        #pragma unroll
        for (int j = 0; j < 8; ++j)
            cb[(half * 8 + j) * CSTR + t * 16 + wl] = f2bf(acc[t][j]);
    asm volatile("s_wait_dscnt 0" ::: "memory");

    // re-read as A-fragments (K = H = 128 -> 4 K-tiles)
    FragB ca[4];
    const unsigned short* crow = cb + wl * CSTR;
    #pragma unroll
    for (int k = 0; k < 4; ++k) {
        const int base = k * 32 + half * 16;
        ca[k].q[0] = *(const u32x4*)(crow + base);
        ca[k].q[1] = *(const u32x4*)(crow + base + 8);
    }

    // ---- phase 2: streaming softmax over q (8 chunks of 64) ----
    v8f zf = {};
    v8f O[8];
    #pragma unroll
    for (int t = 0; t < 8; ++t) O[t] = zf;
    float mrun[8], lrun[8];
    #pragma unroll
    for (int j = 0; j < 8; ++j) { mrun[j] = -1e30f; lrun[j] = 0.f; }

    unsigned short* pb = pbuf + wave * 16 * PSTR;

    for (int c = 0; c < 8; ++c) {
        const int q0 = c * 64;
        v8f s[4];
        #pragma unroll
        for (int t = 0; t < 4; ++t) s[t] = zf;

        // S = ctx_logits @ ql^T   (K = 128)
        #pragma unroll
        for (int t = 0; t < 4; ++t) {
            const unsigned short* qrow =
                qlb + (size_t)(q0 + t * 16 + wl) * H + half * 16;
            #pragma unroll
            for (int k = 0; k < 4; ++k) {
                FragB qb;
                qb.q[0] = *(const u32x4*)(qrow + k * 32);
                qb.q[1] = *(const u32x4*)(qrow + k * 32 + 8);
                s[t] = wmma_bf16(ca[k].v, qb.v, s[t]);
            }
        }
        // mask (column q is lane-resident in D layout)
        #pragma unroll
        for (int t = 0; t < 4; ++t)
            if (mb[q0 + t * 16 + wl] == 0) {
                #pragma unroll
                for (int j = 0; j < 8; ++j) s[t][j] = -1e30f;
            }
        // row max across chunk (16 lanes within each half hold one row)
        float cm[8];
        #pragma unroll
        for (int j = 0; j < 8; ++j)
            cm[j] = fmaxf(fmaxf(s[0][j], s[1][j]), fmaxf(s[2][j], s[3][j]));
        #pragma unroll
        for (int x = 1; x < 16; x <<= 1)
            #pragma unroll
            for (int j = 0; j < 8; ++j)
                cm[j] = fmaxf(cm[j], __shfl_xor(cm[j], x, 32));

        float al[8];
        #pragma unroll
        for (int j = 0; j < 8; ++j) {
            float mn = fmaxf(mrun[j], cm[j]);
            al[j] = __expf(mrun[j] - mn);
            mrun[j] = mn;
        }
        // P = exp(S - m), row sums
        float ls[8];
        #pragma unroll
        for (int j = 0; j < 8; ++j) ls[j] = 0.f;
        #pragma unroll
        for (int t = 0; t < 4; ++t)
            #pragma unroll
            for (int j = 0; j < 8; ++j) {
                float p = __expf(s[t][j] - mrun[j]);
                s[t][j] = p;
                ls[j] += p;
            }
        #pragma unroll
        for (int x = 1; x < 16; x <<= 1)
            #pragma unroll
            for (int j = 0; j < 8; ++j) ls[j] += __shfl_xor(ls[j], x, 32);
        #pragma unroll
        for (int j = 0; j < 8; ++j) lrun[j] = lrun[j] * al[j] + ls[j];
        #pragma unroll
        for (int t = 0; t < 8; ++t)
            #pragma unroll
            for (int j = 0; j < 8; ++j) O[t][j] *= al[j];

        // D-layout -> A-fragment layout via per-wave LDS round-trip
        #pragma unroll
        for (int t = 0; t < 4; ++t)
            #pragma unroll
            for (int j = 0; j < 8; ++j)
                pb[(half * 8 + j) * PSTR + t * 16 + wl] = f2bf(s[t][j]);
        asm volatile("s_wait_dscnt 0" ::: "memory");

        FragB pa[2];
        const unsigned short* prow = pb + wl * PSTR;
        #pragma unroll
        for (int kk = 0; kk < 2; ++kk) {
            const int base = kk * 32 + half * 16;
            pa[kk].q[0] = *(const u32x4*)(prow + base);
            pa[kk].q[1] = *(const u32x4*)(prow + base + 8);
        }
        // O += P @ ql   (B-frag = column h of ql = row h of qlT, contiguous)
        #pragma unroll
        for (int t = 0; t < 8; ++t) {
            const unsigned short* qtr =
                qlTb + (size_t)(t * 16 + wl) * QST + q0 + half * 16;
            #pragma unroll
            for (int kk = 0; kk < 2; ++kk) {
                FragB qb2;
                qb2.q[0] = *(const u32x4*)(qtr + kk * 32);
                qb2.q[1] = *(const u32x4*)(qtr + kk * 32 + 8);
                O[t] = wmma_bf16(pa[kk].v, qb2.v, O[t]);
            }
        }
    }

    // ---- epilogue: normalize, store fp32 ----
    float inv[8];
    #pragma unroll
    for (int j = 0; j < 8; ++j) inv[j] = 1.f / lrun[j];
    #pragma unroll
    for (int t = 0; t < 8; ++t)
        #pragma unroll
        for (int j = 0; j < 8; ++j)
            out[((size_t)b * CTX + m0 + half * 8 + j) * H + t * 16 + wl] =
                O[t][j] * inv[j];
}

// ---------------- host launcher ----------------
extern "C" void kernel_launch(void* const* d_in, const int* in_sizes, int n_in,
                              void* d_out, int out_size, void* d_ws, size_t ws_size,
                              hipStream_t stream) {
    const float* ctx  = (const float*)d_in[0];   // [8,4096,300]
    const float* qst  = (const float*)d_in[1];   // [8,512,300]
    const int*   msk  = (const int*)d_in[2];     // [8,512]
    const float* W    = (const float*)d_in[3];   // [128,300]
    const float* bias = (const float*)d_in[4];   // [128]
    float* out = (float*)d_out;                  // [8,4096,128]

    char* ws = (char*)d_ws;
    const size_t SZ_CTXP = (size_t)Bn * CTX * EP * 2;   // 20,971,520
    const size_t SZ_QSTP = (size_t)Bn * QST * EP * 2;   //  2,621,440
    const size_t SZ_WP   = (size_t)H * EP * 2;          //     81,920
    const size_t SZ_QL   = (size_t)Bn * QST * H * 2;    //  1,048,576
    unsigned short* ctx_pad = (unsigned short*)(ws);
    unsigned short* qst_pad = (unsigned short*)(ws + SZ_CTXP);
    unsigned short* W_pad   = (unsigned short*)(ws + SZ_CTXP + SZ_QSTP);
    unsigned short* ql      = (unsigned short*)(ws + SZ_CTXP + SZ_QSTP + SZ_WP);
    unsigned short* qlT     = (unsigned short*)(ws + SZ_CTXP + SZ_QSTP + SZ_WP + SZ_QL);

    // 1) convert + pad to bf16
    {
        int r = Bn * CTX;
        convert_pad_kernel<<<(r * EP + 255) / 256, 256, 0, stream>>>(ctx, ctx_pad, r);
        r = Bn * QST;
        convert_pad_kernel<<<(r * EP + 255) / 256, 256, 0, stream>>>(qst, qst_pad, r);
        r = H;
        convert_pad_kernel<<<(r * EP + 255) / 256, 256, 0, stream>>>(W, W_pad, r);
    }
    // 2) question logits in two layouts (L2-resident, 2 MB total)
    qst_logits_kernel<<<Bn * (QST / 128), 256, 0, stream>>>(
        qst_pad, W_pad, bias, ql, qlT);
    // 3) fused dense + attention
    fused_attn_kernel<<<Bn * (CTX / 128), 256, 0, stream>>>(
        ctx_pad, W_pad, bias, ql, qlT, msk, out);
}
